// _FactorGCL_83288005804666
// MI455X (gfx1250) — matrix operand order repaired and checked by reading, not compiled, not run on hardware
//
#include <hip/hip_runtime.h>
#include <hip/hip_bf16.h>

// ---------------------------------------------------------------------------
// FactorGCL on MI455X (gfx1250): LayerNorm -> 2x GRU -> hypergraph tail.
// All dense GEMMs use v_wmma_f32_16x16x32_bf16 (wave32 WMMA, f32 accum).
// Each GRU block owns 16 rows and runs the full 64-step recurrence with the
// hidden state resident in LDS; next timestep's x-tile is double-buffered
// into LDS with GLOBAL_LOAD_ASYNC_TO_LDS_B128 (ASYNCcnt).
// ---------------------------------------------------------------------------

#define N_ROWS   4000
#define T_STEPS  64
#define FH       128     // F == H == 128
#define G3       384     // 3*H
#define E_IND    64      // industry count
#define K_PROTO  32      // prototypes

typedef __attribute__((ext_vector_type(16))) __bf16          v16bf;
typedef __attribute__((ext_vector_type(8)))  float           v8f;
typedef __attribute__((ext_vector_type(8)))  unsigned short  ushort8;
typedef __attribute__((ext_vector_type(4)))  unsigned short  ushort4v;

union ABFrag { ushort8 h[2]; v16bf v; };

#if __has_builtin(__builtin_amdgcn_global_load_async_to_lds_b128)
#define HAVE_ASYNC_LDS 1
#else
#define HAVE_ASYNC_LDS 0
#endif

#define GLOBAL_AS __attribute__((address_space(1)))
#define LDS_AS    __attribute__((address_space(3)))

// Pointee type must match the builtin signature exactly:
//   int __attribute__((vector_size(16)))
typedef int v4i_b128 __attribute__((vector_size(16)));

// ---- async copy helpers ----------------------------------------------------
__device__ __forceinline__ void async_copy_b128(const void* gsrc, void* ldst) {
#if HAVE_ASYNC_LDS
    typedef GLOBAL_AS v4i_b128* gvp;
    typedef LDS_AS    v4i_b128* lvp;
    gvp g = reinterpret_cast<gvp>(reinterpret_cast<uintptr_t>(gsrc));
    lvp l = reinterpret_cast<lvp>(
        static_cast<unsigned int>(reinterpret_cast<uintptr_t>(ldst)));
    __builtin_amdgcn_global_load_async_to_lds_b128(g, l, 0, 0);
#else
    (void)gsrc; (void)ldst;
#endif
}
__device__ __forceinline__ void wait_async0() {
#if __has_builtin(__builtin_amdgcn_s_wait_asynccnt)
    __builtin_amdgcn_s_wait_asynccnt(0);
#else
    asm volatile("s_wait_asynccnt 0" ::: "memory");
#endif
}

// ---- scalar conversions ----------------------------------------------------
__device__ __forceinline__ unsigned short f2bf(float f) {
    unsigned int u = __float_as_uint(f);
    unsigned int r = (u + 0x7FFFu + ((u >> 16) & 1u)) >> 16;
    return (unsigned short)r;
}
__device__ __forceinline__ float bf2f(unsigned short u) {
    return __uint_as_float(((unsigned int)u) << 16);
}
__device__ __forceinline__ float sigmoidf_(float v) {
    return 1.0f / (1.0f + __expf(-v));
}
__device__ __forceinline__ float lreluf_(float v) {
    return v > 0.0f ? v : 0.01f * v;
}

// ---- WMMA fragment loaders (layouts per CDNA5 ISA 7.12.2) ------------------
// A (16x32 bf16): lane L<16 holds row M=L, K = {k0..k0+7, k0+16..k0+23};
//                 lane L+16 holds row M=L, K = {k0+8..15, k0+24..31}.
__device__ __forceinline__ v16bf load_a_bf16(const unsigned short* base,
                                             long long ld, int k0, int lane) {
    int m = lane & 15, half = lane >> 4;
    const unsigned short* r = base + (long long)m * ld;
    ABFrag u;
    u.h[0] = *(const ushort8*)(r + k0 + half * 8);
    u.h[1] = *(const ushort8*)(r + k0 + 16 + half * 8);
    return u.v;
}
// B (32x16 bf16, from row-major W[Nout,K], computing X @ W^T):
// lane j<16 -> col n0+j, K = {k0..k0+7, k0+16..k0+23} ; lane j+16 -> col n0+j, K = {k0+8..15, k0+24..31}.
// two 16-byte chunks per lane.
__device__ __forceinline__ v16bf load_b_bf16(const unsigned short* w,
                                             int ldb, int n0, int k0, int lane) {
    int n = n0 + (lane & 15), half = lane >> 4;
    const unsigned short* r = w + (long long)n * ldb + k0 + half * 8;
    ABFrag u;
    u.h[0] = *(const ushort8*)(r);
    u.h[1] = *(const ushort8*)(r + 16);
    return u.v;
}
__device__ __forceinline__ v8f wmma_bf16(v16bf a, v16bf b, v8f c) {
    return __builtin_amdgcn_wmma_f32_16x16x32_bf16(
        /*neg_a=*/false, a, /*neg_b=*/false, b,
        /*c_mod=*/(short)0, c, /*reuse_a=*/false, /*reuse_b=*/false);
}
// C/D 16x16 f32: lane<16 -> col=lane, rows 0..7; lane>=16 -> col=lane-16, rows 8..15
__device__ __forceinline__ void store_c_lds(float* dst, int ldd, int n0,
                                            v8f c, int lane) {
    int col = n0 + (lane & 15);
    int r0  = (lane >> 4) * 8;
    #pragma unroll
    for (int r = 0; r < 8; ++r) dst[(r0 + r) * ldd + col] = c[r];
}

// ---------------------------------------------------------------------------
// Kernel: f32 -> bf16 conversion
// ---------------------------------------------------------------------------
__global__ void cvt_f32_bf16(const float* __restrict__ in,
                             unsigned short* __restrict__ out, int n) {
    int i = blockIdx.x * blockDim.x + threadIdx.x;
    if (i < n) out[i] = f2bf(in[i]);
}

// ---------------------------------------------------------------------------
// Kernel: LayerNorm over F=128, output bf16.  1 wave per row, 8 rows/block.
// ---------------------------------------------------------------------------
__global__ __launch_bounds__(256) void ln_kernel(
    const float* __restrict__ x, const float* __restrict__ g,
    const float* __restrict__ b, unsigned short* __restrict__ out,
    long long rows) {
    long long row = (long long)blockIdx.x * 8 + (threadIdx.x >> 5);
    int lane = threadIdx.x & 31;
    if (row >= rows) return;
    const float* xr = x + row * FH;
    float4 v = ((const float4*)xr)[lane];
    float s  = v.x + v.y + v.z + v.w;
    float s2 = v.x * v.x + v.y * v.y + v.z * v.z + v.w * v.w;
    #pragma unroll
    for (int off = 16; off > 0; off >>= 1) {
        s  += __shfl_xor(s,  off, 32);
        s2 += __shfl_xor(s2, off, 32);
    }
    float mu   = s * (1.0f / FH);
    float var  = s2 * (1.0f / FH) - mu * mu;
    float rstd = rsqrtf(var + 1e-5f);
    int j = lane * 4;
    ushort4v o;
    o.x = f2bf((v.x - mu) * rstd * g[j + 0] + b[j + 0]);
    o.y = f2bf((v.y - mu) * rstd * g[j + 1] + b[j + 1]);
    o.z = f2bf((v.z - mu) * rstd * g[j + 2] + b[j + 2]);
    o.w = f2bf((v.w - mu) * rstd * g[j + 3] + b[j + 3]);
    ((ushort4v*)(out + row * FH))[lane] = o;
}

// ---------------------------------------------------------------------------
// Kernel: one full GRU layer.  Block = 16 rows x 8 waves; h lives in LDS
// across all 64 timesteps.  Wave w owns gate columns [48w, 48w+48).
// x-tiles are async-prefetched into a double-buffered LDS stage.
// ---------------------------------------------------------------------------
__global__ __launch_bounds__(256) void gru_kernel(
    const unsigned short* __restrict__ in_seq,   // (N,T,128) bf16
    const unsigned short* __restrict__ Wih,      // (384,128) bf16
    const unsigned short* __restrict__ Whh,      // (384,128) bf16
    const float* __restrict__ b_ih, const float* __restrict__ b_hh,
    unsigned short* __restrict__ out_seq,        // (N,T,128) bf16 or null
    float* __restrict__ out_last) {              // (N,128) f32 or null
    __shared__ __align__(32) float          xg[16][G3];
    __shared__ __align__(32) float          hg[16][G3];
    __shared__ __align__(32) unsigned short hcur[16][FH];
#if HAVE_ASYNC_LDS
    __shared__ __align__(32) unsigned short xstage[2][16][FH];
#endif

    const int tid  = threadIdx.x;
    const int lane = tid & 31;
    const int wave = tid >> 5;          // 0..7
    const int n0   = blockIdx.x * 16;
    const int c0   = wave * 48;         // first gate column for this wave

#if HAVE_ASYNC_LDS
    // 256 threads x 16B = one full 16x128 bf16 tile per issue
    const int srow = tid >> 4;          // 0..15
    const int schk = (tid & 15) * 8;    // element offset of 16B chunk
    {   // preload t = 0 into buffer 0
        const unsigned short* g =
            in_seq + ((long long)(n0 + srow) * T_STEPS + 0) * FH + schk;
        async_copy_b128(g, &xstage[0][srow][schk]);
    }
#endif
    for (int i = tid; i < 16 * FH; i += 256) ((unsigned short*)hcur)[i] = 0;
#if HAVE_ASYNC_LDS
    wait_async0();
#endif
    __syncthreads();

    for (int t = 0; t < T_STEPS; ++t) {
#if HAVE_ASYNC_LDS
        // kick off async stage of the next timestep while we compute
        if (t + 1 < T_STEPS) {
            const unsigned short* g =
                in_seq + ((long long)(n0 + srow) * T_STEPS + (t + 1)) * FH + schk;
            async_copy_b128(g, &xstage[(t + 1) & 1][srow][schk]);
        }
        const unsigned short* xbase = &xstage[t & 1][0][0];
        const long long xld = FH;
#else
        const unsigned short* xbase =
            in_seq + ((long long)n0 * T_STEPS + t) * FH;
        const long long xld = (long long)T_STEPS * FH;
        if (t + 1 < T_STEPS)
            __builtin_prefetch(xbase + FH +
                               (long long)(lane & 15) * (T_STEPS * FH), 0, 0);
#endif

        // ---- input gates: xg = x_t @ Wih^T  (3 col tiles per wave) ----
        {
            v8f a0 = {}, a1 = {}, a2 = {};
            #pragma unroll
            for (int k0 = 0; k0 < FH; k0 += 32) {
                v16bf a  = load_a_bf16(xbase, xld, k0, lane);
                v16bf b0 = load_b_bf16(Wih, FH, c0,      k0, lane);
                v16bf b1 = load_b_bf16(Wih, FH, c0 + 16, k0, lane);
                v16bf b2 = load_b_bf16(Wih, FH, c0 + 32, k0, lane);
                a0 = wmma_bf16(a, b0, a0);
                a1 = wmma_bf16(a, b1, a1);
                a2 = wmma_bf16(a, b2, a2);
            }
            store_c_lds(&xg[0][0], G3, c0,      a0, lane);
            store_c_lds(&xg[0][0], G3, c0 + 16, a1, lane);
            store_c_lds(&xg[0][0], G3, c0 + 32, a2, lane);
        }
        // ---- hidden gates: hg = h @ Whh^T  (h from LDS) ----
        {
            v8f a0 = {}, a1 = {}, a2 = {};
            #pragma unroll
            for (int k0 = 0; k0 < FH; k0 += 32) {
                v16bf a  = load_a_bf16(&hcur[0][0], FH, k0, lane);
                v16bf b0 = load_b_bf16(Whh, FH, c0,      k0, lane);
                v16bf b1 = load_b_bf16(Whh, FH, c0 + 16, k0, lane);
                v16bf b2 = load_b_bf16(Whh, FH, c0 + 32, k0, lane);
                a0 = wmma_bf16(a, b0, a0);
                a1 = wmma_bf16(a, b1, a1);
                a2 = wmma_bf16(a, b2, a2);
            }
            store_c_lds(&hg[0][0], G3, c0,      a0, lane);
            store_c_lds(&hg[0][0], G3, c0 + 16, a1, lane);
            store_c_lds(&hg[0][0], G3, c0 + 32, a2, lane);
        }
        __syncthreads();

        // ---- fused gate combine: 16*128 = 2048 elems, 8 per thread ----
        #pragma unroll
        for (int i = 0; i < 8; ++i) {
            int idx = tid + i * 256;
            int m = idx >> 7;
            int j = idx & 127;
            float xr = xg[m][j]       + b_ih[j];
            float xz = xg[m][128 + j] + b_ih[128 + j];
            float xn = xg[m][256 + j] + b_ih[256 + j];
            float hr = hg[m][j]       + b_hh[j];
            float hz = hg[m][128 + j] + b_hh[128 + j];
            float hn = hg[m][256 + j] + b_hh[256 + j];
            float r = sigmoidf_(xr + hr);
            float z = sigmoidf_(xz + hz);
            float n = tanhf(xn + r * hn);
            float hprev = bf2f(hcur[m][j]);
            float h = (1.0f - z) * n + z * hprev;
            hcur[m][j] = f2bf(h);
            if (out_seq)
                out_seq[((long long)(n0 + m) * T_STEPS + t) * FH + j] = f2bf(h);
            if (out_last && t == T_STEPS - 1)
                out_last[(long long)(n0 + m) * FH + j] = h;
        }
#if HAVE_ASYNC_LDS
        wait_async0();   // stage for t+1 complete in this wave
#endif
        __syncthreads(); // ...and visible block-wide
    }
}

// ---------------------------------------------------------------------------
// Kernel: generic C[M,Nout] = act(A_bf16[M,K] @ B_bf16[Nout,K]^T + bias)
// block = 16 rows; blockDim = 32 * (Nout/16); act: 0=none 1=lrelu 2=sigmoid
// ---------------------------------------------------------------------------
__global__ void wmma_gemm_nt(const unsigned short* __restrict__ A,
                             const unsigned short* __restrict__ B,
                             const float* __restrict__ bias,
                             float* __restrict__ C,
                             int Nout, int K, int act) {
    int lane = threadIdx.x & 31;
    int wave = threadIdx.x >> 5;
    long long m0 = (long long)blockIdx.x * 16;
    int n0 = wave * 16;
    v8f acc = {};
    for (int k0 = 0; k0 < K; k0 += 32) {
        v16bf a = load_a_bf16(A + m0 * K, K, k0, lane);
        v16bf b = load_b_bf16(B, K, n0, k0, lane);
        acc = wmma_bf16(a, b, acc);
    }
    int col = n0 + (lane & 15);
    int r0  = (lane >> 4) * 8;
    float bb = bias ? bias[col] : 0.0f;
    #pragma unroll
    for (int r = 0; r < 8; ++r) {
        float v = acc[r] + bb;
        if (act == 1)      v = lreluf_(v);
        else if (act == 2) v = sigmoidf_(v);
        C[(m0 + r0 + r) * Nout + col] = v;
    }
}

// ---------------------------------------------------------------------------
// Hypergraph tail kernels (bandwidth-trivial)
// ---------------------------------------------------------------------------
__global__ void prior_row_deg(const int* __restrict__ Hp, float* __restrict__ dninv) {
    int n = blockIdx.x * blockDim.x + threadIdx.x;
    if (n >= N_ROWS) return;
    int s = 0;
    for (int e = 0; e < E_IND; ++e) s += Hp[(long long)n * E_IND + e];
    dninv[n] = s > 0 ? 1.0f / (float)s : 0.0f;
}
__global__ void prior_col_deg(const int* __restrict__ Hp, float* __restrict__ deinv) {
    int e = threadIdx.x;
    if (e >= E_IND) return;
    int s = 0;
    for (int n = 0; n < N_ROWS; ++n) s += Hp[(long long)n * E_IND + e];
    deinv[e] = s > 0 ? 1.0f / (float)s : 0.0f;
}
__global__ void prior_m_kernel(const int* __restrict__ Hp, const float* __restrict__ t1,
                               const float* __restrict__ deinv, float* __restrict__ m2) {
    int e = blockIdx.x, h = threadIdx.x;
    float acc = 0.0f;
    for (int n = 0; n < N_ROWS; ++n) {
        int w = Hp[(long long)n * E_IND + e];
        if (w) acc += (float)w * t1[(long long)n * FH + h];
    }
    m2[(long long)e * FH + h] = acc * deinv[e];
}
__global__ void prior_out_kernel(const int* __restrict__ Hp, const float* __restrict__ m2,
                                 const float* __restrict__ dninv, const float* __restrict__ bp,
                                 const float* __restrict__ es,
                                 float* __restrict__ ep, float* __restrict__ er) {
    int n = blockIdx.x, h = threadIdx.x;
    float acc = 0.0f;
    for (int e = 0; e < E_IND; ++e)
        acc += (float)Hp[(long long)n * E_IND + e] * m2[(long long)e * FH + h];
    float v = lreluf_(acc * dninv[n] + bp[h]);
    ep[(long long)n * FH + h] = v;
    er[(long long)n * FH + h] = es[(long long)n * FH + h] - v;
}
__global__ void soft_row_deg(const float* __restrict__ B, float* __restrict__ dninvs) {
    int n = blockIdx.x * blockDim.x + threadIdx.x;
    if (n >= N_ROWS) return;
    float s = 0.0f;
    for (int k = 0; k < K_PROTO; ++k) s += B[(long long)n * K_PROTO + k];
    dninvs[n] = s > 0.0f ? rsqrtf(s) : 0.0f;
}
__global__ void soft_col_deg(const float* __restrict__ B, float* __restrict__ deinv) {
    int k = threadIdx.x;
    if (k >= K_PROTO) return;
    float s = 0.0f;
    for (int n = 0; n < N_ROWS; ++n) s += B[(long long)n * K_PROTO + k];
    deinv[k] = s > 0.0f ? 1.0f / s : 0.0f;
}
__global__ void soft_m_kernel(const float* __restrict__ B, const float* __restrict__ s1,
                              const float* __restrict__ deinv, float* __restrict__ m3) {
    int k = blockIdx.x, h = threadIdx.x;
    float acc = 0.0f;
    for (int n = 0; n < N_ROWS; ++n)
        acc += B[(long long)n * K_PROTO + k] * s1[(long long)n * FH + h];
    m3[(long long)k * FH + h] = acc * deinv[k];
}
__global__ void soft_out_kernel(const float* __restrict__ B, const float* __restrict__ m3,
                                const float* __restrict__ dninvs, const float* __restrict__ er,
                                float* __restrict__ eh, float* __restrict__ eres) {
    int n = blockIdx.x, h = threadIdx.x;
    float acc = 0.0f;
    for (int k = 0; k < K_PROTO; ++k)
        acc += B[(long long)n * K_PROTO + k] * m3[(long long)k * FH + h];
    float v = lreluf_(acc * dninvs[n]);
    eh[(long long)n * FH + h] = v;
    eres[(long long)n * FH + h] = er[(long long)n * FH + h] - v;
}
__global__ __launch_bounds__(128) void final_kernel(
    const float* __restrict__ ep, const float* __restrict__ eh,
    const float* __restrict__ ea, const float* __restrict__ Wf,
    const float* __restrict__ bf, float* __restrict__ out) {
    int n = blockIdx.x, h = threadIdx.x;
    float v = Wf[h]       * ep[(long long)n * FH + h] +
              Wf[128 + h] * eh[(long long)n * FH + h] +
              Wf[256 + h] * ea[(long long)n * FH + h];
    #pragma unroll
    for (int off = 16; off > 0; off >>= 1) v += __shfl_xor(v, off, 32);
    __shared__ float red[4];
    if ((threadIdx.x & 31) == 0) red[threadIdx.x >> 5] = v;
    __syncthreads();
    if (threadIdx.x == 0)
        out[n] = red[0] + red[1] + red[2] + red[3] + bf[0];
}

// ---------------------------------------------------------------------------
// Host launcher
// ---------------------------------------------------------------------------
extern "C" void kernel_launch(void* const* d_in, const int* in_sizes, int n_in,
                              void* d_out, int out_size, void* d_ws, size_t ws_size,
                              hipStream_t stream) {
    const float* x     = (const float*)d_in[0];
    const int*   Hp    = (const int*)  d_in[1];
    const float* ln_g  = (const float*)d_in[2];
    const float* ln_b  = (const float*)d_in[3];
    const float* W_ih0 = (const float*)d_in[4];
    const float* W_hh0 = (const float*)d_in[5];
    const float* b_ih0 = (const float*)d_in[6];
    const float* b_hh0 = (const float*)d_in[7];
    const float* W_ih1 = (const float*)d_in[8];
    const float* W_hh1 = (const float*)d_in[9];
    const float* b_ih1 = (const float*)d_in[10];
    const float* b_hh1 = (const float*)d_in[11];
    const float* Wp    = (const float*)d_in[12];
    const float* bp    = (const float*)d_in[13];
    const float* proto = (const float*)d_in[14];
    const float* Ws    = (const float*)d_in[15];
    const float* bs    = (const float*)d_in[16];
    const float* Wa    = (const float*)d_in[17];
    const float* ba    = (const float*)d_in[18];
    const float* Wf    = (const float*)d_in[19];
    const float* bfp   = (const float*)d_in[20];
    float* out = (float*)d_out;

    // ---- workspace layout (bump allocated; tail aliases dead xn arena) ----
    const size_t SEQ_BYTES = (size_t)N_ROWS * T_STEPS * FH * 2;   // 65,536,000
    char* base = (char*)d_ws;
    unsigned short* xn = (unsigned short*)base;                   // arena A
    unsigned short* h1 = (unsigned short*)(base + SEQ_BYTES);     // arena B
    char* w = base + 2 * SEQ_BYTES;                               // weights
    unsigned short* wih0b = (unsigned short*)(w);            //  98304 B
    unsigned short* whh0b = (unsigned short*)(w +  98304);
    unsigned short* wih1b = (unsigned short*)(w + 196608);
    unsigned short* whh1b = (unsigned short*)(w + 294912);
    unsigned short* wpb   = (unsigned short*)(w + 393216);   // 32768
    unsigned short* protob= (unsigned short*)(w + 425984);   //  8192
    unsigned short* wsb   = (unsigned short*)(w + 434176);   // 32768
    unsigned short* wab   = (unsigned short*)(w + 466944);   // 32768

    // tail buffers alias arena A (xn is dead after GRU layer 0)
    char* q = base;
    float*          es    = (float*)q;          q += 2048000;
    unsigned short* esb   = (unsigned short*)q; q += 1024000;
    float*          t1    = (float*)q;          q += 2048000;
    float*          dninv = (float*)q;          q += 16384;
    float*          deinv = (float*)q;          q += 256;
    float*          m2    = (float*)q;          q += 32768;
    float*          ep    = (float*)q;          q += 2048000;
    float*          er    = (float*)q;          q += 2048000;
    unsigned short* erb   = (unsigned short*)q; q += 1024000;
    float*          beta  = (float*)q;          q += 512000;
    float*          s1    = (float*)q;          q += 2048000;
    float*          dn2   = (float*)q;          q += 16384;
    float*          de2   = (float*)q;          q += 256;
    float*          m3    = (float*)q;          q += 16384;
    float*          eh    = (float*)q;          q += 2048000;
    float*          eres  = (float*)q;          q += 2048000;
    unsigned short* eresb = (unsigned short*)q; q += 1024000;
    float*          ea    = (float*)q;          q += 2048000;

    // ---- 1. weights -> bf16 ----
    cvt_f32_bf16<<<(49152 + 255) / 256, 256, 0, stream>>>(W_ih0, wih0b, 49152);
    cvt_f32_bf16<<<(49152 + 255) / 256, 256, 0, stream>>>(W_hh0, whh0b, 49152);
    cvt_f32_bf16<<<(49152 + 255) / 256, 256, 0, stream>>>(W_ih1, wih1b, 49152);
    cvt_f32_bf16<<<(49152 + 255) / 256, 256, 0, stream>>>(W_hh1, whh1b, 49152);
    cvt_f32_bf16<<<(16384 + 255) / 256, 256, 0, stream>>>(Wp,    wpb,   16384);
    cvt_f32_bf16<<<( 4096 + 255) / 256, 256, 0, stream>>>(proto, protob, 4096);
    cvt_f32_bf16<<<(16384 + 255) / 256, 256, 0, stream>>>(Ws,    wsb,   16384);
    cvt_f32_bf16<<<(16384 + 255) / 256, 256, 0, stream>>>(Wa,    wab,   16384);

    // ---- 2. LayerNorm (f32 -> bf16), 256000 rows ----
    ln_kernel<<<(N_ROWS * T_STEPS) / 8, 256, 0, stream>>>(
        x, ln_g, ln_b, xn, (long long)N_ROWS * T_STEPS);

    // ---- 3/4. GRU layers (WMMA, h resident in LDS, async x staging) ----
    gru_kernel<<<N_ROWS / 16, 256, 0, stream>>>(
        xn, wih0b, whh0b, b_ih0, b_hh0, h1, nullptr);
    gru_kernel<<<N_ROWS / 16, 256, 0, stream>>>(
        h1, wih1b, whh1b, b_ih1, b_hh1, nullptr, es);

    // ---- 5. e_s -> bf16; t1 = e_s @ Wp^T ----
    cvt_f32_bf16<<<(N_ROWS * FH + 255) / 256, 256, 0, stream>>>(es, esb, N_ROWS * FH);
    wmma_gemm_nt<<<N_ROWS / 16, 256, 0, stream>>>(esb, wpb, nullptr, t1, FH, FH, 0);

    // ---- 6. prior hconv ----
    prior_row_deg<<<(N_ROWS + 255) / 256, 256, 0, stream>>>(Hp, dninv);
    prior_col_deg<<<1, E_IND, 0, stream>>>(Hp, deinv);
    prior_m_kernel<<<E_IND, FH, 0, stream>>>(Hp, t1, deinv, m2);
    prior_out_kernel<<<N_ROWS, FH, 0, stream>>>(Hp, m2, dninv, bp, es, ep, er);

    // ---- 7. beta = sigmoid(e_r @ proto^T); s1 = e_r @ Ws^T + bs ----
    cvt_f32_bf16<<<(N_ROWS * FH + 255) / 256, 256, 0, stream>>>(er, erb, N_ROWS * FH);
    wmma_gemm_nt<<<N_ROWS / 16, 64, 0, stream>>>(erb, protob, nullptr, beta, K_PROTO, FH, 2);
    wmma_gemm_nt<<<N_ROWS / 16, 256, 0, stream>>>(erb, wsb, bs, s1, FH, FH, 0);

    // ---- 8. soft hconv ----
    soft_row_deg<<<(N_ROWS + 255) / 256, 256, 0, stream>>>(beta, dn2);
    soft_col_deg<<<1, K_PROTO, 0, stream>>>(beta, de2);
    soft_m_kernel<<<K_PROTO, FH, 0, stream>>>(beta, s1, de2, m3);
    soft_out_kernel<<<N_ROWS, FH, 0, stream>>>(beta, m3, dn2, er, eh, eres);

    // ---- 9. e_alpha = lrelu(e_res @ Wa^T + ba) ----
    cvt_f32_bf16<<<(N_ROWS * FH + 255) / 256, 256, 0, stream>>>(eres, eresb, N_ROWS * FH);
    wmma_gemm_nt<<<N_ROWS / 16, 256, 0, stream>>>(eresb, wab, ba, ea, FH, FH, 1);

    // ---- 10. final projection ----
    final_kernel<<<N_ROWS, FH, 0, stream>>>(ep, eh, ea, Wf, bfp, out);

    (void)in_sizes; (void)n_in; (void)out_size; (void)ws_size;
}
